// KANLayer_73761768341660
// MI455X (gfx1250) — compile-verified, matrix operand order checked by reading, not run on hardware
//
#include <hip/hip_runtime.h>
#include <hip/hip_bf16.h>
#include <math.h>

typedef __attribute__((ext_vector_type(16))) _Float16 v16h;
typedef __attribute__((ext_vector_type(8)))  float    v8f;
typedef __attribute__((ext_vector_type(4)))  int      i4;

#define BATCH   32768
#define DDIM    256
#define KIN     10
#define PDIM    (DDIM * KIN)      // 2560
#define NCHUNK  (PDIM / 32)       // 80 k-chunks of 32
#define MTILES  2
#define ROWS    (16 * MTILES)     // 32 batch rows per wave
#define XSTRIDE 260               // row-major x tile stride: 260 % 64 == 4 -> bank-skewed

#if defined(__has_builtin)
#  if __has_builtin(__builtin_amdgcn_tanhf)
#    define HAS_TANH 1
#  endif
#  if __has_builtin(__builtin_amdgcn_global_load_async_to_lds_b128) && \
      __has_builtin(__builtin_amdgcn_s_wait_asynccnt)
#    define HAS_ASYNC 1
#  endif
#endif

typedef __attribute__((address_space(1))) i4 gi4;   // global 16B vector
typedef __attribute__((address_space(3))) i4 li4;   // LDS 16B vector

__device__ __forceinline__ float fast_tanh(float v) {
#ifdef HAS_TANH
    return __builtin_amdgcn_tanhf(v);          // v_tanh_f32 (1 trans op)
#else
    float e = __expf(2.0f * v);                // fallback: exp + rcp
    return 1.0f - 2.0f * __builtin_amdgcn_rcpf(e + 1.0f);
#endif
}

// ---------------------------------------------------------------------------
// Prep 1: fold Wbig[p,n] = sum_j w2[d, p%10, j] * wo1[d*10+j, n]  (d = p/10)
// directly into WMMA B-fragment order: for k-chunk kc, lane L holds 16 f16
// values: col n = L&15, k-locals {0..7,16..23} (L<16) / {8..15,24..31} (L>=16).
// ---------------------------------------------------------------------------
__global__ __launch_bounds__(128) void kan_prep_w(const float* __restrict__ w2,
                                                  const float* __restrict__ wo1,
                                                  _Float16* __restrict__ wfrag) {
    int t = blockIdx.x * blockDim.x + threadIdx.x;   // one thread per (kc, lane)
    if (t >= PDIM) return;                           // 80*32 == 2560
    int kc   = t >> 5;
    int lane = t & 31;
    int half = lane >> 4;
    int n    = lane & 15;
    _Float16 vals[16];
#pragma unroll
    for (int i = 0; i < 16; ++i) {
        int kl = (i < 8 ? i : i + 8) + (half ? 8 : 0);
        int p  = kc * 32 + kl;
        int d  = p / KIN;
        int kk = p - d * KIN;
        float s = 0.0f;
        if (n < KIN) {
            const float* w2row = w2 + (size_t)(d * KIN + kk) * KIN;
            const float* worow = wo1 + (size_t)(d * KIN) * KIN + n;
#pragma unroll
            for (int j = 0; j < KIN; ++j)
                s += w2row[j] * worow[(size_t)j * KIN];
        }
        vals[i] = (_Float16)s;
    }
#pragma unroll
    for (int i = 0; i < 16; ++i)
        wfrag[(size_t)t * 16 + i] = vals[i];
}

// Prep 2: c[n] = bo1[n] + sum_p b2[p] * wo1[p*10+n]  (padded to 16 with zeros)
__global__ __launch_bounds__(32) void kan_prep_c(const float* __restrict__ b2,
                                                 const float* __restrict__ wo1,
                                                 const float* __restrict__ bo1,
                                                 float* __restrict__ cbuf) {
    int n = threadIdx.x;
    if (n >= 16) return;
    float s = 0.0f;
    if (n < KIN) {
        s = bo1[n];
        for (int p = 0; p < PDIM; ++p)
            s += b2[p] * wo1[(size_t)p * KIN + n];
    }
    cbuf[n] = s;
}

// ---------------------------------------------------------------------------
// Main: one wave32 per 32-row batch tile (two 16x16 WMMA tiles, shared B frag).
//   H[b, p] = tanh(x[b, p/10] * w1[p] + b1[p]) computed inline into A frags,
//   acc[16x16] += A(16x32) x B(32x16) over 80 chunks (V_WMMA_F32_16X16X32_F16),
//   out = tanh(acc + c) @ wo2 + bo2 via ds_add_f32 reduction.
// ---------------------------------------------------------------------------
__global__ __launch_bounds__(32) void kan_main(const float* __restrict__ x,
                                               const float* __restrict__ w1,
                                               const float* __restrict__ b1,
                                               const float* __restrict__ wo2,
                                               const float* __restrict__ bo2,
                                               const _Float16* __restrict__ wfrag,
                                               const float* __restrict__ cbuf,
                                               float* __restrict__ out) {
    __shared__ float xs[ROWS * XSTRIDE];   // row-major x tile, bank-skewed rows
    __shared__ float red[ROWS];

    const int lane = threadIdx.x;
    const int b0   = blockIdx.x * ROWS;
    const int half = lane >> 4;
    const int arow = lane & 15;     // A-matrix row (M) for this lane
    const int ncol = lane & 15;     // C-matrix column (N) for this lane

    // stage x tile: xs[r*260 + c] = x[b0+r, c], 16B chunks per lane
#ifdef HAS_ASYNC
    for (int idx = lane; idx < ROWS * (DDIM / 4); idx += 32) {
        int r = idx >> 6;                 // DDIM/4 == 64 chunks per row
        int c = (idx & 63) << 2;
        const float* g = x + (size_t)(b0 + r) * DDIM + c;
        float*       l = &xs[r * XSTRIDE + c];
        __builtin_amdgcn_global_load_async_to_lds_b128((gi4*)g, (li4*)l, 0, 0);
    }
    __builtin_amdgcn_s_wait_asynccnt(0);
#else
    for (int idx = lane; idx < ROWS * (DDIM / 4); idx += 32) {
        int r = idx >> 6;
        int c = (idx & 63) << 2;
        *(float4*)&xs[r * XSTRIDE + c] =
            *(const float4*)(x + (size_t)(b0 + r) * DDIM + c);
    }
#endif
    if (lane < ROWS) red[lane] = 0.0f;
    __syncthreads();

    const float cn  = cbuf[ncol];
    const float won = (ncol < KIN) ? wo2[ncol] : 0.0f;

    v8f acc0 = {};
    v8f acc1 = {};
    const int koff0 = half ? 8 : 0;    // first run of 8 k-locals
    const int koff1 = half ? 24 : 16;  // second run of 8 k-locals

    for (int kc = 0; kc < NCHUNK; ++kc) {
        const int p0 = kc * 32 + koff0;
        const int p1 = kc * 32 + koff1;

        // B fragment: 16 contiguous halves per lane (pre-swizzled by prep)
        const v16h bfrag = *(const v16h*)(wfrag + ((size_t)(kc * 32 + lane)) * 16);
        if (kc + 1 < NCHUNK)
            __builtin_prefetch(wfrag + ((size_t)((kc + 1) * 32 + lane)) * 16, 0, 0);

        alignas(16) float wv0[8], bv0[8], wv1[8], bv1[8];
        *(float4*)(wv0)     = *(const float4*)(w1 + p0);
        *(float4*)(wv0 + 4) = *(const float4*)(w1 + p0 + 4);
        *(float4*)(bv0)     = *(const float4*)(b1 + p0);
        *(float4*)(bv0 + 4) = *(const float4*)(b1 + p0 + 4);
        *(float4*)(wv1)     = *(const float4*)(w1 + p1);
        *(float4*)(wv1 + 4) = *(const float4*)(w1 + p1 + 4);
        *(float4*)(bv1)     = *(const float4*)(b1 + p1);
        *(float4*)(bv1 + 4) = *(const float4*)(b1 + p1 + 4);

        // A fragments for both 16-row tiles (exactly one tanh per H element)
        v16h a0{}, a1{};
#pragma unroll
        for (int i = 0; i < 8; ++i) {
            int p = p0 + i;
            int d = p / KIN;
            float xv0 = xs[arow * XSTRIDE + d];
            float xv1 = xs[(arow + 16) * XSTRIDE + d];
            a0[i] = (_Float16)fast_tanh(fmaf(xv0, wv0[i], bv0[i]));
            a1[i] = (_Float16)fast_tanh(fmaf(xv1, wv0[i], bv0[i]));
        }
#pragma unroll
        for (int i = 0; i < 8; ++i) {
            int p = p1 + i;
            int d = p / KIN;
            float xv0 = xs[arow * XSTRIDE + d];
            float xv1 = xs[(arow + 16) * XSTRIDE + d];
            a0[8 + i] = (_Float16)fast_tanh(fmaf(xv0, wv1[i], bv1[i]));
            a1[8 + i] = (_Float16)fast_tanh(fmaf(xv1, wv1[i], bv1[i]));
        }

        acc0 = __builtin_amdgcn_wmma_f32_16x16x32_f16(
            false, a0, false, bfrag, (short)0, acc0, false, false);
        acc1 = __builtin_amdgcn_wmma_f32_16x16x32_f16(
            false, a1, false, bfrag, (short)0, acc1, false, false);
    }

    // epilogue: u = tanh(acc + c[n]); out_row += u * wo2[n]
#pragma unroll
    for (int i = 0; i < 8; ++i) {
        int r = half * 8 + i;                    // C layout: lanes<16 rows 0..7, lanes>=16 rows 8..15
        float u0 = fast_tanh(acc0[i] + cn);
        float u1 = fast_tanh(acc1[i] + cn);
        float c0 = (ncol < KIN) ? u0 * won : 0.0f;
        float c1 = (ncol < KIN) ? u1 * won : 0.0f;
        atomicAdd(&red[r],      c0);             // ds_add_f32
        atomicAdd(&red[r + 16], c1);
    }
    __syncthreads();
    out[b0 + lane] = red[lane] + bo2[0];         // all 32 lanes store one row
}

extern "C" void kernel_launch(void* const* d_in, const int* in_sizes, int n_in,
                              void* d_out, int out_size, void* d_ws, size_t ws_size,
                              hipStream_t stream) {
    const float* x   = (const float*)d_in[0];
    const float* w1  = (const float*)d_in[1];
    const float* b1  = (const float*)d_in[2];
    const float* w2  = (const float*)d_in[3];
    const float* b2  = (const float*)d_in[4];
    const float* wo1 = (const float*)d_in[5];
    const float* bo1 = (const float*)d_in[6];
    const float* wo2 = (const float*)d_in[7];
    const float* bo2 = (const float*)d_in[8];

    _Float16* wfrag = (_Float16*)d_ws;                                  // 2560*16 f16 = 80 KB
    float*    cbuf  = (float*)((char*)d_ws + (size_t)PDIM * 16 * 2);    // +64 B

    kan_prep_w<<<(PDIM + 127) / 128, 128, 0, stream>>>(w2, wo1, wfrag);
    kan_prep_c<<<1, 32, 0, stream>>>(b2, wo1, bo1, cbuf);
    kan_main<<<BATCH / ROWS, 32, 0, stream>>>(x, w1, b1, wo2, bo2, wfrag, cbuf,
                                              (float*)d_out);
}
